// DeepSeekBlock_82068235092054
// MI455X (gfx1250) — compile-verified, matrix operand order checked by reading, not run on hardware
//
#include <hip/hip_runtime.h>

// ---------------------------------------------------------------------------
// DeepSeek-style block for MI455X (gfx1250): bf16 WMMA everywhere, TDM staging.
//   B=8 S=2048 D=1024 H=16 HD=64 E=32 K=8 F=1024   (BS = 16384 tokens)
// ---------------------------------------------------------------------------

typedef __attribute__((ext_vector_type(16))) __bf16    v16bf;
typedef __attribute__((ext_vector_type(8)))  float     v8f;
typedef __attribute__((ext_vector_type(4)))  float     f32x4;
typedef __attribute__((ext_vector_type(4)))  unsigned  uvec4;
typedef __attribute__((ext_vector_type(4)))  unsigned  u32x4;
typedef __attribute__((ext_vector_type(8)))  int       i32x8;
typedef __attribute__((ext_vector_type(4)))  int       i32x4;

#define BS   16384
#define DMODEL 1024

#if defined(__has_builtin)
#if __has_builtin(__builtin_amdgcn_tensor_load_to_lds)
#define HAVE_TDM 1
#endif
#endif
#ifndef HAVE_TDM
#define HAVE_TDM 0
#endif

// ---------------- small helpers ----------------
__device__ __forceinline__ unsigned short bf16bits(float f) {
  unsigned u = __float_as_uint(f);
  unsigned r = (u + 0x7FFFu + ((u >> 16) & 1u)) >> 16;
  return (unsigned short)r;
}
__device__ __forceinline__ float bf16tof(unsigned short s) {
  return __uint_as_float(((unsigned)s) << 16);
}

// Padded LDS row mapping: row = 1024 bf16, +16B pad after each 1024B.
// Matches TDM pad_enable with pad_interval=7 (1024B) pad_amount=3 (16B).
#define ROWB 2080
__device__ __forceinline__ unsigned pofs(int m, int k) {
  return (unsigned)(m * ROWB + (k << 1) + ((k >> 9) << 4));
}

// A fragment (16x32 bf16): lane L -> m = L&15, khalf = L>>4.
// lane holds k = k0..k0+7 and k0+16..k0+23 where k0 = ktile*32 + khalf*8.
__device__ __forceinline__ v16bf load_a(const char* base, int m, int k0) {
  union { uvec4 u[2]; v16bf v; } r;
  r.u[0] = *(const uvec4*)(base + pofs(m, k0));
  r.u[1] = *(const uvec4*)(base + pofs(m, k0 + 16));
  return r.v;
}
// B fragment from pre-swizzled weights: tile = 32 lanes x 16 bf16 contiguous (1KB)
__device__ __forceinline__ v16bf load_b(const unsigned short* tileBase, int lane) {
  union { uvec4 u[2]; v16bf v; } r;
  const uvec4* p = (const uvec4*)(tileBase + (lane << 4));
  r.u[0] = p[0];
  r.u[1] = p[1];
  return r.v;
}
__device__ __forceinline__ v8f wmma_bf16(v16bf a, v16bf b, v8f c) {
  return __builtin_amdgcn_wmma_f32_16x16x32_bf16(false, a, false, b, (short)0, c,
                                                 false, false);
}
__device__ __forceinline__ void wait_tensor0() {
#if defined(__has_builtin)
#if __has_builtin(__builtin_amdgcn_s_wait_tensorcnt)
  __builtin_amdgcn_s_wait_tensorcnt((short)0);
#else
  asm volatile("s_wait_tensorcnt 0" ::: "memory");
#endif
#else
  asm volatile("s_wait_tensorcnt 0" ::: "memory");
#endif
}

// ---------------- TDM descriptors ----------------
#if HAVE_TDM
// 2D tile load: rows x 1024 bf16 from row-major [BS][1024] buffer into padded LDS
__device__ __forceinline__ void tdm_load_tile2d(unsigned ldsOff, const void* gaddr,
                                                unsigned rows) {
  unsigned long long ga = (unsigned long long)gaddr;
  u32x4 g0; i32x8 g1; i32x4 gz; i32x8 gz8;
  gz[0] = gz[1] = gz[2] = gz[3] = 0;
  gz8[0] = gz8[1] = gz8[2] = gz8[3] = gz8[4] = gz8[5] = gz8[6] = gz8[7] = 0;
  g0[0] = 1u;                                   // count=1, no gather
  g0[1] = ldsOff;                               // lds_addr
  g0[2] = (unsigned)ga;                         // global_addr lo
  g0[3] = (unsigned)((ga >> 32) & 0x01FFFFFFu) | (2u << 30);  // hi | type=2
  g1[0] = (int)((1u << 16) | (1u << 20) | (7u << 22) | (3u << 25)); // 2B, pad
  g1[1] = (int)(1024u << 16);                   // tensor_dim0 = 1024
  g1[2] = (int)(16384u << 16);                  // tensor_dim1 = BS
  g1[3] = (int)(1024u << 16);                   // tile_dim0 = 1024
  g1[4] = (int)rows;                            // tile_dim1
  g1[5] = 1024;                                 // tensor_dim0_stride
  g1[6] = 0; g1[7] = 0;
  __builtin_amdgcn_tensor_load_to_lds(g0, g1, gz, gz, gz8, 0);
}
// gather-mode: 16 rows (16-bit indices) of 1024 bf16 each into padded LDS
__device__ __forceinline__ void tdm_gather16(unsigned ldsOff, const void* gaddr,
                                             const int* tokp) {
  unsigned long long ga = (unsigned long long)gaddr;
  u32x4 g0; i32x8 g1; i32x4 g2, g3; i32x8 gz8;
  gz8[0] = gz8[1] = gz8[2] = gz8[3] = gz8[4] = gz8[5] = gz8[6] = gz8[7] = 0;
  g0[0] = 1u | (1u << 31);                      // count=1 | gather_mode (16-bit idx)
  g0[1] = ldsOff;
  g0[2] = (unsigned)ga;
  g0[3] = (unsigned)((ga >> 32) & 0x01FFFFFFu) | (2u << 30);
  g1[0] = (int)((1u << 16) | (1u << 20) | (7u << 22) | (3u << 25));
  g1[1] = (int)(1024u << 16);
  g1[2] = (int)(16384u << 16);
  g1[3] = (int)(1024u << 16);
  g1[4] = 16;                                   // tile_dim1 = #indices
  g1[5] = 1024;
  g1[6] = 0; g1[7] = 0;
#pragma unroll
  for (int i = 0; i < 4; ++i)
    g2[i] = (tokp[2 * i] & 0xFFFF) | ((tokp[2 * i + 1] & 0xFFFF) << 16);
#pragma unroll
  for (int i = 0; i < 4; ++i)
    g3[i] = (tokp[8 + 2 * i] & 0xFFFF) | ((tokp[9 + 2 * i] & 0xFFFF) << 16);
  __builtin_amdgcn_tensor_load_to_lds(g0, g1, g2, g3, gz8, 0);
}
#endif

// ---------------- weight swizzle: fp32 [E*K][N] -> bf16 B-fragment tiles -------
// tile(e,kt,nt) holds 32 lanes x 16 bf16; lane L: n = nt*16 + (L&15),
// element j: k = kt*32 + (j&7) + (j&8 ? 16 : 0) + (L>>4)*8
extern "C" __global__ __launch_bounds__(256) void k_swz(
    const float* __restrict__ src, unsigned short* __restrict__ dst, int Erep,
    int Kdim, int Ndim) {
  size_t t = (size_t)blockIdx.x * 256 + threadIdx.x;
  int j = (int)(t & 15);
  int lane = (int)((t >> 4) & 31);
  size_t tl = t >> 9;
  int ntiles = Ndim >> 4, ktiles = Kdim >> 5;
  int nt = (int)(tl % ntiles);
  size_t r = tl / ntiles;
  int kt = (int)(r % ktiles);
  int e = (int)(r / ktiles);
  if (e >= Erep) return;
  int hi = lane >> 4, n = lane & 15;
  int k = kt * 32 + (j & 7) + ((j & 8) ? 16 : 0) + hi * 8;
  int ng = nt * 16 + n;
  float v = src[((size_t)e * Kdim + k) * Ndim + ng];
  dst[t] = bf16bits(v);
}

extern "C" __global__ void k_zero32(int* p) {
  if (threadIdx.x < 32) p[threadIdx.x] = 0;
}

// ---------------- LayerNorm -> bf16 (used for LN1 and LN2) --------------------
extern "C" __global__ __launch_bounds__(256) void k_ln_bf16(
    const float* __restrict__ x, const float* __restrict__ sc,
    const float* __restrict__ bi, unsigned short* __restrict__ out) {
  __shared__ float red[8], red2[8];
  const int tid = threadIdx.x;
  const size_t row = blockIdx.x;
  f32x4 v = ((const f32x4*)(x + row * DMODEL))[tid];
  float s = v[0] + v[1] + v[2] + v[3];
  float q = v[0] * v[0] + v[1] * v[1] + v[2] * v[2] + v[3] * v[3];
#pragma unroll
  for (int m = 16; m >= 1; m >>= 1) {
    s += __shfl_xor(s, m, 32);
    q += __shfl_xor(q, m, 32);
  }
  if ((tid & 31) == 0) { red[tid >> 5] = s; red2[tid >> 5] = q; }
  __syncthreads();
  if (tid == 0) {
    float a = 0.f, b = 0.f;
    for (int i = 0; i < 8; ++i) { a += red[i]; b += red2[i]; }
    red[0] = a; red2[0] = b;
  }
  __syncthreads();
  float mean = red[0] * (1.f / 1024.f);
  float var = red2[0] * (1.f / 1024.f) - mean * mean;
  float rs = rsqrtf(var + 1e-5f);
  int d0 = tid * 4;
#pragma unroll
  for (int i = 0; i < 4; ++i) {
    float y = (v[i] - mean) * rs * sc[d0 + i] + bi[d0 + i];
    out[row * DMODEL + d0 + i] = bf16bits(y);
  }
}

// ---------------- QKV GEMM: [BS x 1024] @ [1024 x 1024] x3, WMMA bf16 --------
extern "C" __global__ __launch_bounds__(256) void k_qkv(
    const unsigned short* __restrict__ xln, const unsigned short* __restrict__ wqs,
    const unsigned short* __restrict__ wks, const unsigned short* __restrict__ wvs,
    const float* __restrict__ bq, const float* __restrict__ bk,
    const float* __restrict__ bv, unsigned short* __restrict__ qb,
    unsigned short* __restrict__ kb, unsigned short* __restrict__ vb) {
  extern __shared__ char smem[];   // 32 rows x ROWB
  const int tid = threadIdx.x, lane = tid & 31, w = tid >> 5;
  const int m0 = blockIdx.x * 32;
  const int nb = blockIdx.y, mat = nb >> 3;
  const int nt = (nb & 7) * 8 + w;  // 0..63 within matrix
  const unsigned short* wmatS = (mat == 0) ? wqs : (mat == 1) ? wks : wvs;
  const float* bias = (mat == 0) ? bq : (mat == 1) ? bk : bv;
  unsigned short* ob = (mat == 0) ? qb : (mat == 1) ? kb : vb;
#if HAVE_TDM
  if (w == 0) {
    unsigned xsOff = (unsigned)(size_t)(void*)smem;
    tdm_load_tile2d(xsOff, xln + (size_t)m0 * DMODEL, 32u);
    wait_tensor0();
  }
#else
  for (int c = tid; c < 4096; c += 256) {
    int r = c >> 7, kk = (c & 127) * 8;
    *(uvec4*)(smem + pofs(r, kk)) =
        *(const uvec4*)(xln + (size_t)(m0 + r) * DMODEL + kk);
  }
#endif
  __syncthreads();
  const int hi = lane >> 4, nn = lane & 15;
  v8f acc0 = {0.f, 0.f, 0.f, 0.f, 0.f, 0.f, 0.f, 0.f};
  v8f acc1 = acc0;
#pragma unroll 4
  for (int kt = 0; kt < 32; ++kt) {
    int k0 = kt * 32 + hi * 8;
    v16bf a0 = load_a(smem, nn, k0);
    v16bf a1 = load_a(smem, 16 + nn, k0);
    v16bf b = load_b(wmatS + ((size_t)(kt * 64 + nt) << 9), lane);
    acc0 = wmma_bf16(a0, b, acc0);
    acc1 = wmma_bf16(a1, b, acc1);
  }
  int ng = nt * 16 + nn;
  float bs = bias[ng];
#pragma unroll
  for (int r = 0; r < 8; ++r) {
    int mr = hi ? 8 + r : r;
    ob[(size_t)(m0 + mr) * DMODEL + ng] = bf16bits(acc0[r] + bs);
    ob[(size_t)(m0 + 16 + mr) * DMODEL + ng] = bf16bits(acc1[r] + bs);
  }
}

// ---------------- per-token attention (16 heads x 16 heads, K=64) + residual --
extern "C" __global__ __launch_bounds__(256) void k_attn(
    const unsigned short* __restrict__ qb, const unsigned short* __restrict__ kbuf,
    const unsigned short* __restrict__ vb, const float* __restrict__ hidden,
    float* __restrict__ out) {
  __shared__ unsigned short qls[8][1024];
  __shared__ unsigned short kls[8][1024];
  __shared__ unsigned short vls[8][1024];
  __shared__ unsigned short pls[8][264];
  const int tid = threadIdx.x, lane = tid & 31, w = tid >> 5;
  const size_t t = (size_t)blockIdx.x * 8 + w;
  {
    const uvec4* qs = (const uvec4*)(qb + t * DMODEL);
    const uvec4* ks = (const uvec4*)(kbuf + t * DMODEL);
    const uvec4* vs = (const uvec4*)(vb + t * DMODEL);
    uvec4* qd = (uvec4*)qls[w];
    uvec4* kd = (uvec4*)kls[w];
    uvec4* vd = (uvec4*)vls[w];
    for (int i = lane; i < 128; i += 32) { qd[i] = qs[i]; kd[i] = ks[i]; vd[i] = vs[i]; }
  }
  __syncthreads();
  const int hi = lane >> 4, nn = lane & 15;
  // scores[h][g] = sum_d q[h][d]*k[g][d] / 8
  v8f sc = {0.f, 0.f, 0.f, 0.f, 0.f, 0.f, 0.f, 0.f};
#pragma unroll
  for (int c = 0; c < 2; ++c) {
    union { unsigned short s[16]; v16bf v; } A, Bm;
#pragma unroll
    for (int j = 0; j < 16; ++j) {
      int ko = (j & 7) + ((j & 8) ? 16 : 0) + hi * 8 + c * 32;
      A.s[j] = qls[w][nn * 64 + ko];   // A: m = nn (head), k = d
      Bm.s[j] = kls[w][nn * 64 + ko];  // B: n = nn (g), k = d  -> k[g][d]
    }
    sc = wmma_bf16(A.v, Bm.v, sc);
  }
  // row-wise softmax (rows live in regs, cols across 16-lane halves)
#pragma unroll
  for (int r = 0; r < 8; ++r) {
    float x = sc[r] * 0.125f;
    float mx = x;
#pragma unroll
    for (int m = 8; m >= 1; m >>= 1) mx = fmaxf(mx, __shfl_xor(mx, m, 32));
    float e = __expf(x - mx);
    float sm = e;
#pragma unroll
    for (int m = 8; m >= 1; m >>= 1) sm += __shfl_xor(sm, m, 32);
    float p = e / sm;
    int mrow = hi ? 8 + r : r;
    pls[w][mrow * 16 + nn] = bf16bits(p);
  }
  __syncthreads();
  // ctx[h][d] = sum_g p[h][g]*v[g][d]  (K padded 16 -> 32 with zeros)
#pragma unroll
  for (int c = 0; c < 4; ++c) {
    union { unsigned short s[16]; v16bf v; } A, Bv;
#pragma unroll
    for (int j = 0; j < 16; ++j) {
      int ko = (j & 7) + ((j & 8) ? 16 : 0) + hi * 8;
      A.s[j] = (ko < 16) ? pls[w][nn * 16 + ko] : (unsigned short)0;
      Bv.s[j] = (ko < 16) ? vls[w][ko * 64 + c * 16 + nn] : (unsigned short)0;
    }
    v8f ct = {0.f, 0.f, 0.f, 0.f, 0.f, 0.f, 0.f, 0.f};
    ct = wmma_bf16(A.v, Bv.v, ct);
#pragma unroll
    for (int r = 0; r < 8; ++r) {
      int h = hi ? 8 + r : r;
      size_t idx = t * DMODEL + h * 64 + c * 16 + nn;
      out[idx] = ct[r] + hidden[idx];   // h1 = hidden + ctx
    }
  }
}

// ---------------- router: logits -> top-8 -> per-expert (token, weight) lists -
extern "C" __global__ __launch_bounds__(256) void k_router(
    const unsigned short* __restrict__ x2, const float* __restrict__ rw,
    const float* __restrict__ rb, int* __restrict__ cnt, int* __restrict__ tokl,
    float* __restrict__ wl) {
  __shared__ float part[256];
  __shared__ float lg[32];
  const int tid = threadIdx.x;
  const size_t row = blockIdx.x;
  int e = tid >> 3, p = tid & 7;
  float a = 0.f;
  for (int d = p * 128; d < p * 128 + 128; ++d)
    a += bf16tof(x2[row * DMODEL + d]) * rw[(size_t)d * 32 + e];
  part[tid] = a;
  __syncthreads();
  if (p == 0) {
    float s = 0.f;
#pragma unroll
    for (int i = 0; i < 8; ++i) s += part[(e << 3) + i];
    lg[e] = s + rb[e];
  }
  __syncthreads();
  if (tid == 0) {
    unsigned used = 0;
    int sel[8]; float lv[8];
    for (int j = 0; j < 8; ++j) {
      float best = -1e30f; int bidx = 0;
      for (int i = 0; i < 32; ++i)
        if (!((used >> i) & 1u) && lg[i] > best) { best = lg[i]; bidx = i; }
      used |= 1u << bidx; sel[j] = bidx; lv[j] = best;
    }
    float mx = lv[0], ssum = 0.f, wv[8];
    for (int j = 0; j < 8; ++j) { wv[j] = __expf(lv[j] - mx); ssum += wv[j]; }
    for (int j = 0; j < 8; ++j) {
      int ex = sel[j];
      int slot = atomicAdd(&cnt[ex], 1);
      tokl[(size_t)ex * BS + slot] = (int)row;
      wl[(size_t)ex * BS + slot] = wv[j] / ssum;
    }
  }
}

// ---------------- fused MoE FFN per (expert, 32-token tile) -------------------
extern "C" __global__ __launch_bounds__(256) void k_moe(
    const unsigned short* __restrict__ x2, const unsigned short* __restrict__ w1s,
    const float* __restrict__ b1, const unsigned short* __restrict__ w2s,
    const float* __restrict__ b2, const int* __restrict__ cnt,
    const int* __restrict__ tokl, const float* __restrict__ wl,
    float* __restrict__ out) {
  extern __shared__ char smem[];   // xs: 32*ROWB, hs: 32*ROWB
  char* xs = smem;
  char* hs = smem + 32 * ROWB;
  __shared__ int tok[32];
  __shared__ float gw[32];
  const int e = blockIdx.y;
  const int n = cnt[e];
  const int base = blockIdx.x * 32;
  if (base >= n) return;
  const int tid = threadIdx.x, lane = tid & 31, w = tid >> 5;
  if (tid < 32) {
    int ii = base + tid;
    tok[tid] = (ii < n) ? tokl[(size_t)e * BS + ii] : 0;
    gw[tid] = (ii < n) ? wl[(size_t)e * BS + ii] : 0.f;
  }
  __syncthreads();
#if HAVE_TDM
  if (w == 0) {
    unsigned xsOff = (unsigned)(size_t)(void*)xs;
    tdm_gather16(xsOff, x2, tok);
    tdm_gather16(xsOff + 16u * ROWB, x2, tok + 16);
    wait_tensor0();
  }
#else
  for (int c = tid; c < 4096; c += 256) {
    int r = c >> 7, kk = (c & 127) * 8;
    *(uvec4*)(xs + pofs(r, kk)) =
        *(const uvec4*)(x2 + (size_t)tok[r] * DMODEL + kk);
  }
#endif
  __syncthreads();
  const int hi = lane >> 4, nn = lane & 15;
  const unsigned short* w1e = w1s + (size_t)e * 1048576;
  const unsigned short* w2e = w2s + (size_t)e * 1048576;
  // ---- phase 1: h = gelu(xs @ w1[e] + b1[e])  (wave owns 128 F-cols) ----
  v8f acc[2][8] = {};
#pragma unroll 2
  for (int kt = 0; kt < 32; ++kt) {
    int k0 = kt * 32 + hi * 8;
    v16bf a0 = load_a(xs, nn, k0);
    v16bf a1 = load_a(xs, 16 + nn, k0);
    const unsigned short* tb = w1e + ((size_t)(kt * 64 + w * 8) << 9);
#pragma unroll
    for (int i = 0; i < 8; ++i) {
      v16bf b = load_b(tb + ((size_t)i << 9), lane);
      acc[0][i] = wmma_bf16(a0, b, acc[0][i]);
      acc[1][i] = wmma_bf16(a1, b, acc[1][i]);
    }
  }
#pragma unroll
  for (int i = 0; i < 8; ++i) {
    int ng = (w * 8 + i) * 16 + nn;
    float bb = b1[(size_t)e * 1024 + ng];
#pragma unroll
    for (int r = 0; r < 8; ++r) {
#pragma unroll
      for (int mt = 0; mt < 2; ++mt) {
        float xv = acc[mt][i][r] + bb;
        float g = 0.5f * xv * (1.f + erff(xv * 0.70710678f));
        int mr = mt * 16 + (hi ? 8 + r : r);
        *(unsigned short*)(hs + pofs(mr, ng)) = bf16bits(g);
      }
    }
  }
  __syncthreads();
  // ---- phase 2: out += gate * (h @ w2[e] + b2[e]) ----
  v8f z8 = {0.f, 0.f, 0.f, 0.f, 0.f, 0.f, 0.f, 0.f};
#pragma unroll
  for (int mt = 0; mt < 2; ++mt)
#pragma unroll
    for (int i = 0; i < 8; ++i) acc[mt][i] = z8;
#pragma unroll 2
  for (int kt = 0; kt < 32; ++kt) {
    int k0 = kt * 32 + hi * 8;
    v16bf a0 = load_a(hs, nn, k0);
    v16bf a1 = load_a(hs, 16 + nn, k0);
    const unsigned short* tb = w2e + ((size_t)(kt * 64 + w * 8) << 9);
#pragma unroll
    for (int i = 0; i < 8; ++i) {
      v16bf b = load_b(tb + ((size_t)i << 9), lane);
      acc[0][i] = wmma_bf16(a0, b, acc[0][i]);
      acc[1][i] = wmma_bf16(a1, b, acc[1][i]);
    }
  }
#pragma unroll
  for (int i = 0; i < 8; ++i) {
    int ng = (w * 8 + i) * 16 + nn;
    float bb = b2[(size_t)e * 1024 + ng];
#pragma unroll
    for (int r = 0; r < 8; ++r) {
#pragma unroll
      for (int mt = 0; mt < 2; ++mt) {
        int mr = mt * 16 + (hi ? 8 + r : r);
        float val = (acc[mt][i][r] + bb) * gw[mr];
        atomicAdd(&out[(size_t)tok[mr] * DMODEL + ng], val);
      }
    }
  }
}

// ---------------- workspace layout (bytes) ------------------------------------
static const size_t XLN_OFF = 0;                          // BS*D bf16
static const size_t X2_OFF  = XLN_OFF + 33554432;
static const size_t QB_OFF  = X2_OFF + 33554432;
static const size_t KB_OFF  = QB_OFF + 33554432;
static const size_t VB_OFF  = KB_OFF + 33554432;
static const size_t WQS_OFF = VB_OFF + 33554432;          // 1024*1024 bf16
static const size_t WKS_OFF = WQS_OFF + 2097152;
static const size_t WVS_OFF = WKS_OFF + 2097152;
static const size_t W1S_OFF = WVS_OFF + 2097152;          // 32*1024*1024 bf16
static const size_t W2S_OFF = W1S_OFF + 67108864;
static const size_t CNT_OFF = W2S_OFF + 67108864;         // 32 ints
static const size_t TOK_OFF = CNT_OFF + 256;              // 32*16384 ints
static const size_t WL_OFF  = TOK_OFF + 2097152;          // 32*16384 floats

extern "C" void kernel_launch(void* const* d_in, const int* in_sizes, int n_in,
                              void* d_out, int out_size, void* d_ws,
                              size_t ws_size, hipStream_t stream) {
  (void)in_sizes; (void)n_in; (void)out_size; (void)ws_size;
  const float* hidden = (const float*)d_in[0];
  const float* ln1s = (const float*)d_in[1];
  const float* ln1b = (const float*)d_in[2];
  const float* wq = (const float*)d_in[3];
  const float* bq = (const float*)d_in[4];
  const float* wk = (const float*)d_in[5];
  const float* bk = (const float*)d_in[6];
  const float* wv = (const float*)d_in[7];
  const float* bv = (const float*)d_in[8];
  const float* ln2s = (const float*)d_in[9];
  const float* ln2b = (const float*)d_in[10];
  const float* rw = (const float*)d_in[11];
  const float* rb = (const float*)d_in[12];
  const float* w1 = (const float*)d_in[13];
  const float* b1 = (const float*)d_in[14];
  const float* w2 = (const float*)d_in[15];
  const float* b2 = (const float*)d_in[16];
  float* out = (float*)d_out;
  char* ws = (char*)d_ws;

  unsigned short* xln = (unsigned short*)(ws + XLN_OFF);
  unsigned short* x2 = (unsigned short*)(ws + X2_OFF);
  unsigned short* qb = (unsigned short*)(ws + QB_OFF);
  unsigned short* kb = (unsigned short*)(ws + KB_OFF);
  unsigned short* vb = (unsigned short*)(ws + VB_OFF);
  unsigned short* wqs = (unsigned short*)(ws + WQS_OFF);
  unsigned short* wks = (unsigned short*)(ws + WKS_OFF);
  unsigned short* wvs = (unsigned short*)(ws + WVS_OFF);
  unsigned short* w1s = (unsigned short*)(ws + W1S_OFF);
  unsigned short* w2s = (unsigned short*)(ws + W2S_OFF);
  int* cnt = (int*)(ws + CNT_OFF);
  int* tokl = (int*)(ws + TOK_OFF);
  float* wl = (float*)(ws + WL_OFF);

  // allow >64KB dynamic LDS for the WMMA tile kernels (idempotent; errors ignored)
  (void)hipFuncSetAttribute((const void*)k_qkv,
                            hipFuncAttributeMaxDynamicSharedMemorySize, 32 * ROWB);
  (void)hipFuncSetAttribute((const void*)k_moe,
                            hipFuncAttributeMaxDynamicSharedMemorySize,
                            2 * 32 * ROWB);

  // one-time (per call) weight bf16 swizzle into B-fragment tile order
  k_swz<<<4096, 256, 0, stream>>>(wq, wqs, 1, 1024, 1024);
  k_swz<<<4096, 256, 0, stream>>>(wk, wks, 1, 1024, 1024);
  k_swz<<<4096, 256, 0, stream>>>(wv, wvs, 1, 1024, 1024);
  k_swz<<<131072, 256, 0, stream>>>(w1, w1s, 32, 1024, 1024);
  k_swz<<<131072, 256, 0, stream>>>(w2, w2s, 32, 1024, 1024);
  k_zero32<<<1, 32, 0, stream>>>(cnt);

  k_ln_bf16<<<BS, 256, 0, stream>>>(hidden, ln1s, ln1b, xln);
  k_qkv<<<dim3(512, 24), 256, 32 * ROWB, stream>>>(xln, wqs, wks, wvs, bq, bk, bv,
                                                   qb, kb, vb);
  k_attn<<<2048, 256, 0, stream>>>(qb, kb, vb, hidden, out);   // out = h1
  k_ln_bf16<<<BS, 256, 0, stream>>>(out, ln2s, ln2b, x2);
  k_router<<<BS, 256, 0, stream>>>(x2, rw, rb, cnt, tokl, wl);
  k_moe<<<dim3(512, 32), 256, 2 * 32 * ROWB, stream>>>(x2, w1s, b1, w2s, b2, cnt,
                                                       tokl, wl, out);
}
